// GraphBlock_63780264345902
// MI455X (gfx1250) — compile-verified, matrix operand order checked by reading, not run on hardware
//
#include <hip/hip_runtime.h>
#include <hip/hip_bf16.h>
#include <math.h>

#define ND 128
#define WT_STRIDE 136   // padded row stride for transposed W in LDS (16B aligned, bank-staggered)
#define ROWS_PB 64      // rows of X per block (4 row-tiles of 16)
#define NEG_SLOPE 0.2f
#define LN_EPS 1e-5f

typedef __attribute__((ext_vector_type(16))) __bf16 v16bf;
typedef __attribute__((ext_vector_type(8)))  __bf16 v8bf;
typedef __attribute__((ext_vector_type(8)))  float  v8f;

// ---------- order-preserving float<->int for atomicMax on signed int ----------
__device__ __forceinline__ int f2o(float f) {
    int i = __float_as_int(f);
    return i >= 0 ? i : (i ^ 0x7FFFFFFF);
}
__device__ __forceinline__ float o2f(int o) {
    return __int_as_float(o >= 0 ? o : (o ^ 0x7FFFFFFF));
}
__device__ __forceinline__ float gelu_exact(float v) {
    return 0.5f * v * (1.0f + erff(v * 0.70710678118654752f));
}

// ------------------------------- fills ---------------------------------------
__global__ void fill_f32(float* p, float v, long n) {
    long i = (long)blockIdx.x * blockDim.x + threadIdx.x;
    if (i < n) p[i] = v;
}
__global__ void fill_i32(int* p, int v, long n) {
    long i = (long)blockIdx.x * blockDim.x + threadIdx.x;
    if (i < n) p[i] = v;
}

// ------------------------- degree / loop weight ------------------------------
__global__ void deg_kernel(const long long* __restrict__ dst, const float* __restrict__ ew,
                           float* deg, float* wsum, int E) {
    int e = blockIdx.x * blockDim.x + threadIdx.x;
    if (e >= E) return;
    int d = (int)dst[e];
    atomicAdd(&deg[d], 1.0f);
    atomicAdd(&wsum[d], ew[e]);
}
__global__ void loopw_kernel(float* wsum, const float* deg, int n) {
    int i = blockIdx.x * blockDim.x + threadIdx.x;
    if (i < n) wsum[i] = wsum[i] / fmaxf(deg[i], 1.0f);
}

// ---------------------- ce = dot(lin_edge_w, att_edge) -----------------------
__global__ void ce_kernel(const float* __restrict__ lew, const float* __restrict__ ae, float* out) {
    int lane = threadIdx.x; // 32 threads
    float s = 0.f;
    for (int i = lane; i < ND; i += 32) s += lew[i] * ae[i];
    #pragma unroll
    for (int off = 16; off; off >>= 1) s += __shfl_xor(s, off, 32);
    if (lane == 0) out[0] = s;
}

// ----------------------------- WMMA GEMM -------------------------------------
// H[N,128] = X[N,128] @ Wm[128,128]; 64 rows per block (4 row-tiles), 8 waves
// = 8 col tiles. W is stored TRANSPOSED in LDS; B fragments are hoisted into
// registers once and reused across all 4 row tiles (16 WMMAs per wave).
__global__ __launch_bounds__(256) void gemm_wmma(const float* __restrict__ X,
                                                 const float* __restrict__ Wm,
                                                 float* __restrict__ H, long Nn) {
    __shared__ __bf16 sWt[ND * WT_STRIDE];  // ~34 KB, sWt[n*WT_STRIDE + k] = W[k][n]
    __shared__ __bf16 sA[ROWS_PB * ND];     //  16 KB
    const int tid = threadIdx.x;
    const long row0 = (long)blockIdx.x * ROWS_PB;

    // transpose-fill W (coalesced global reads, once per block, amortized over 4 tiles)
    for (int i = tid * 4; i < ND * ND; i += 256 * 4) {
        const int k = i >> 7, n = i & (ND - 1);
        float4 v = *(const float4*)&Wm[i];
        sWt[(n + 0) * WT_STRIDE + k] = (__bf16)v.x;
        sWt[(n + 1) * WT_STRIDE + k] = (__bf16)v.y;
        sWt[(n + 2) * WT_STRIDE + k] = (__bf16)v.z;
        sWt[(n + 3) * WT_STRIDE + k] = (__bf16)v.w;
    }
    // A-tile fill (row-clamped loads for the ragged tail)
    for (int i = tid * 4; i < ROWS_PB * ND; i += 256 * 4) {
        long r = row0 + (i >> 7);
        if (r >= Nn) r = Nn - 1;
        float4 v = *(const float4*)&X[r * ND + (i & (ND - 1))];
        sA[i + 0] = (__bf16)v.x;
        sA[i + 1] = (__bf16)v.y;
        sA[i + 2] = (__bf16)v.z;
        sA[i + 3] = (__bf16)v.w;
    }
    __syncthreads();

    const int wave = tid >> 5;
    const int lane = tid & 31;
    const int half = lane >> 4;   // K-half selector per ISA A/B layout
    const int q    = lane & 15;   // row (A) / col (B,D) index
    const int col0 = wave * 16;
    const int brow = (col0 + q) * WT_STRIDE;

    // hoist B fragments for all 4 K-steps into registers (reused by all row tiles)
    v16bf bfrag[4];
    #pragma unroll
    for (int kk = 0; kk < 4; ++kk) {
        const int ka = kk * 32 + half * 8;
        v8bf b0 = *(const v8bf*)&sWt[brow + ka];
        v8bf b1 = *(const v8bf*)&sWt[brow + ka + 16];
        bfrag[kk] = __builtin_shufflevector(b0, b1, 0, 1, 2, 3, 4, 5, 6, 7,
                                                    8, 9, 10, 11, 12, 13, 14, 15);
    }

    #pragma unroll
    for (int t = 0; t < 4; ++t) {
        const long rowt = row0 + t * 16;
        const int arow = (t * 16 + q) * ND;
        v8f c = {0.f, 0.f, 0.f, 0.f, 0.f, 0.f, 0.f, 0.f};
        #pragma unroll
        for (int kk = 0; kk < 4; ++kk) {
            const int ka = kk * 32 + half * 8;
            v8bf a0 = *(const v8bf*)&sA[arow + ka];
            v8bf a1 = *(const v8bf*)&sA[arow + ka + 16];
            v16bf a = __builtin_shufflevector(a0, a1, 0, 1, 2, 3, 4, 5, 6, 7,
                                                      8, 9, 10, 11, 12, 13, 14, 15);
            c = __builtin_amdgcn_wmma_f32_16x16x32_bf16(false, a, false, bfrag[kk],
                                                        (short)0, c, false, false);
        }
        float* bp = H + ((rowt + half * 8) * (long)ND + col0 + q);
        if (rowt + 16 <= Nn) {           // uniform fast path: branch-free 8 stores
            #pragma unroll
            for (int r = 0; r < 8; ++r) bp[r * ND] = c[r];
        } else if (rowt < Nn) {          // ragged tail tile
            #pragma unroll
            for (int r = 0; r < 8; ++r)
                if (rowt + half * 8 + r < Nn) bp[r * ND] = c[r];
        }
    }
}

// ---------------- alpha_s / alpha_d: one wave per node -----------------------
__global__ void alpha_kernel(const float* __restrict__ H, const float* __restrict__ asrc,
                             const float* __restrict__ adst, float* __restrict__ als,
                             float* __restrict__ ald, int n) {
    long t = (long)blockIdx.x * blockDim.x + threadIdx.x;
    long node = t >> 5;
    int lane = threadIdx.x & 31;
    if (node >= n) return;
    float4 hv = ((const float4*)(H + node * ND))[lane];
    float4 s4 = ((const float4*)asrc)[lane];
    float4 d4 = ((const float4*)adst)[lane];
    float s = hv.x * s4.x + hv.y * s4.y + hv.z * s4.z + hv.w * s4.w;
    float d = hv.x * d4.x + hv.y * d4.y + hv.z * d4.z + hv.w * d4.w;
    #pragma unroll
    for (int off = 16; off; off >>= 1) {
        s += __shfl_xor(s, off, 32);
        d += __shfl_xor(d, off, 32);
    }
    if (lane == 0) { als[node] = s; ald[node] = d; }
}

// ------------------- logits + segment max (self-loops appended) --------------
__global__ void logits_kernel(const long long* __restrict__ src, const long long* __restrict__ dst,
                              const float* __restrict__ ew, const float* __restrict__ loopw,
                              const float* __restrict__ als, const float* __restrict__ ald,
                              const float* __restrict__ cep, float* __restrict__ logits,
                              int* __restrict__ m, int E, int M) {
    int e = blockIdx.x * blockDim.x + threadIdx.x;
    if (e >= M) return;
    int s, d; float w;
    if (e < E) { s = (int)src[e]; d = (int)dst[e]; w = ew[e]; }
    else       { s = d = e - E;  w = loopw[s]; }
    float lg = als[s] + ald[d] + cep[0] * w;
    lg = lg > 0.f ? lg : NEG_SLOPE * lg;
    logits[e] = lg;
    atomicMax(&m[d], f2o(lg));
}

// --------------------------- p = exp(l - m); z = sum -------------------------
__global__ void pz_kernel(const long long* __restrict__ dst, float* __restrict__ logits,
                          const int* __restrict__ m, float* __restrict__ z, int E, int M) {
    int e = blockIdx.x * blockDim.x + threadIdx.x;
    if (e >= M) return;
    int d = (e < E) ? (int)dst[e] : e - E;
    float pp = expf(logits[e] - o2f(m[d]));
    logits[e] = pp;
    atomicAdd(&z[d], pp);
}

// ------------------- scatter: one wave per edge, f32 atomics -----------------
__global__ void scatter_kernel(const long long* __restrict__ src, const long long* __restrict__ dst,
                               const float* __restrict__ p, const float* __restrict__ z,
                               const float* __restrict__ H, float* __restrict__ out,
                               int E, long M) {
    long t = (long)blockIdx.x * blockDim.x + threadIdx.x;
    long e = t >> 5;
    int lane = threadIdx.x & 31;
    if (e >= M) return;
    int s, d;
    if (e < E) { s = (int)src[e]; d = (int)dst[e]; }
    else       { s = d = (int)(e - E); }
    const float* hrow = H + (long)s * ND + lane * 4;
    __builtin_prefetch(hrow, 0, 3);            // global_prefetch_b8: start h[src] gather early
    float coef = p[e] / z[d];
    float4 hv = *(const float4*)hrow;
    float* o = out + (long)d * ND + lane * 4;
    atomicAdd(o + 0, hv.x * coef);
    atomicAdd(o + 1, hv.y * coef);
    atomicAdd(o + 2, hv.z * coef);
    atomicAdd(o + 3, hv.w * coef);
}

// ----------------------- layer 0 epilogue: gelu(acc+bias) --------------------
__global__ void gelu_bias_kernel(float* buf, const float* __restrict__ bias, long n) {
    long i = (long)blockIdx.x * blockDim.x + threadIdx.x;
    if (i >= n) return;
    buf[i] = gelu_exact(buf[i] + bias[i & (ND - 1)]);
}

// ------- layer 1 epilogue: y = LN(x + gelu(acc+bias)); wave per node ---------
__global__ void final_ln_kernel(float* __restrict__ out, const float* __restrict__ x,
                                const float* __restrict__ bias, const float* __restrict__ gamma,
                                const float* __restrict__ beta, int n) {
    long t = (long)blockIdx.x * blockDim.x + threadIdx.x;
    long node = t >> 5;
    int lane = threadIdx.x & 31;
    if (node >= n) return;
    float4 v  = ((const float4*)(out + node * ND))[lane];
    float4 xv = ((const float4*)(x   + node * ND))[lane];
    float4 b4 = ((const float4*)bias)[lane];
    v.x = xv.x + gelu_exact(v.x + b4.x);
    v.y = xv.y + gelu_exact(v.y + b4.y);
    v.z = xv.z + gelu_exact(v.z + b4.z);
    v.w = xv.w + gelu_exact(v.w + b4.w);
    float s = v.x + v.y + v.z + v.w;
    #pragma unroll
    for (int off = 16; off; off >>= 1) s += __shfl_xor(s, off, 32);
    float mu = s * (1.0f / ND);
    float dx = v.x - mu, dy = v.y - mu, dz = v.z - mu, dw = v.w - mu;
    float vs = dx * dx + dy * dy + dz * dz + dw * dw;
    #pragma unroll
    for (int off = 16; off; off >>= 1) vs += __shfl_xor(vs, off, 32);
    float rstd = rsqrtf(vs * (1.0f / ND) + LN_EPS);
    float4 g4 = ((const float4*)gamma)[lane];
    float4 e4 = ((const float4*)beta)[lane];
    float4 o;
    o.x = dx * rstd * g4.x + e4.x;
    o.y = dy * rstd * g4.y + e4.y;
    o.z = dz * rstd * g4.z + e4.z;
    o.w = dw * rstd * g4.w + e4.w;
    ((float4*)(out + node * ND))[lane] = o;
}

// =============================================================================
extern "C" void kernel_launch(void* const* d_in, const int* in_sizes, int n_in,
                              void* d_out, int out_size, void* d_ws, size_t ws_size,
                              hipStream_t stream) {
    const float*     x     = (const float*)d_in[0];
    const long long* ei    = (const long long*)d_in[1]; // [2,E] int64
    const float*     ew    = (const float*)d_in[2];
    const float*     W     = (const float*)d_in[3];     // [2,128,128]
    const float*     asr   = (const float*)d_in[4];     // [2,128]
    const float*     adt   = (const float*)d_in[5];
    const float*     lew   = (const float*)d_in[6];
    const float*     ae    = (const float*)d_in[7];
    const float*     bias  = (const float*)d_in[8];
    const float*     gamma = (const float*)d_in[9];
    const float*     beta  = (const float*)d_in[10];

    const int N = in_sizes[0] / ND;       // 100000
    const int E = in_sizes[1] / 2;        // 1600000
    const int M = E + N;

    const long long* src = ei;
    const long long* dst = ei + E;

    // ---- workspace carve-out ----
    float* H   = (float*)d_ws;            // N*128
    float* B   = H + (size_t)N * ND;      // N*128 (layer-0 output / layer-1 input)
    float* P   = B + (size_t)N * ND;      // M (logits, then p)
    float* deg = P + (size_t)M;           // N
    float* lw  = deg + N;                 // N
    float* als = lw + N;                  // N
    float* ald = als + N;                 // N
    float* z   = ald + N;                 // N
    int*   m   = (int*)(z + N);           // N
    float* ce  = (float*)(m + N);         // 1

    const int T = 256;
    auto gb = [](long n, int t) { return (unsigned)((n + t - 1) / t); };

    // degree + mean loop weight
    fill_f32<<<gb(2L * N, T), T, 0, stream>>>(deg, 0.f, 2L * N); // deg + lw contiguous
    deg_kernel<<<gb(E, T), T, 0, stream>>>(dst, ew, deg, lw, E);
    loopw_kernel<<<gb(N, T), T, 0, stream>>>(lw, deg, N);

    for (int l = 0; l < 2; ++l) {
        const float* Wl  = W   + (size_t)l * ND * ND;
        const float* Xin = (l == 0) ? x : B;
        float*       acc = (l == 0) ? B : (float*)d_out;

        ce_kernel<<<1, 32, 0, stream>>>(lew + l * ND, ae + l * ND, ce);
        gemm_wmma<<<gb(N, ROWS_PB), T, 0, stream>>>(Xin, Wl, H, N);
        alpha_kernel<<<gb((long)N * 32, T), T, 0, stream>>>(H, asr + l * ND, adt + l * ND,
                                                            als, ald, N);
        fill_f32<<<gb(N, T), T, 0, stream>>>(z, 0.f, N);
        fill_i32<<<gb(N, T), T, 0, stream>>>(m, (int)0x80000000, N);
        fill_f32<<<gb((long)N * ND, T), T, 0, stream>>>(acc, 0.f, (long)N * ND);

        logits_kernel<<<gb(M, T), T, 0, stream>>>(src, dst, ew, lw, als, ald, ce, P, m, E, M);
        pz_kernel<<<gb(M, T), T, 0, stream>>>(dst, P, m, z, E, M);
        scatter_kernel<<<gb((long)M * 32, T), T, 0, stream>>>(src, dst, P, z, H, acc, E, M);

        if (l == 0)
            gelu_bias_kernel<<<gb((long)N * ND, T), T, 0, stream>>>(B, bias, (long)N * ND);
        else
            final_ln_kernel<<<gb((long)N * 32, T), T, 0, stream>>>((float*)d_out, x,
                                                                   bias + ND, gamma, beta, N);
    }
}
